// GroupQueryAttention_7112465842394
// MI455X (gfx1250) — matrix operand order repaired and checked by reading, not compiled, not run on hardware
//
#include <hip/hip_runtime.h>
#include <math.h>

// Problem constants (match reference)
#define BB    2
#define NSEQ  2048
#define DIN   2048
#define DOUT  2048
#define NH    32
#define NG    8
#define HDIM  64
#define KVD   (NG * HDIM)   // 512

// GEMM macro-tile config
#define TM 128
#define TN 64
#define KC 32

typedef __attribute__((ext_vector_type(16))) __bf16 bf16x16;
typedef __attribute__((ext_vector_type(8)))  float  floatx8;
typedef int v4i __attribute__((ext_vector_type(4)));

union Frag {
    bf16x16        v;
    unsigned int   u[8];
    unsigned short s[16];
};

__device__ __forceinline__ unsigned short f2bf(float f) {
    unsigned int u = __float_as_uint(f);
    u += 0x7FFFu + ((u >> 16) & 1u);   // round to nearest even
    return (unsigned short)(u >> 16);
}

// A fragment (16x32 bf16, MxK): lane holds row m=lane&15; element e -> K = e + 16*half.
__device__ __forceinline__ void load_a_frag(Frag& fr, const unsigned short* rowptr, int half) {
    const unsigned int* p32 = reinterpret_cast<const unsigned int*>(rowptr);
#pragma unroll
    for (int p = 0; p < 8; ++p) {
        int idx = p + 8 * half;
        fr.u[p] = p32[idx];
    }
}

// B fragment (32x16 bf16, KxN): lane holds col n=lane&15; element e -> K = e + 16*half.
// Source is the TRANSPOSED matrix (row n holds all K) -> contiguous u32 pairs.
__device__ __forceinline__ void load_bt_frag(Frag& fr, const unsigned short* rowptr, int half) {
    const unsigned int* p32 = reinterpret_cast<const unsigned int*>(rowptr);
#pragma unroll
    for (int p = 0; p < 8; ++p) fr.u[p] = p32[p + 8 * half];
}

// ---------------- CDNA5 async global->LDS copy (ASYNCcnt path) with sync fallback ----------------
#if defined(__has_builtin)
#  if __has_builtin(__builtin_amdgcn_global_load_async_to_lds_b128)
#    define HAVE_ASYNC_LDS 1
#  endif
#endif

#ifdef HAVE_ASYNC_LDS
typedef __attribute__((address_space(3))) v4i v4i_lds;
#endif

__device__ __forceinline__ void cp16_to_lds(unsigned short* lds, const unsigned short* g) {
#ifdef HAVE_ASYNC_LDS
    __builtin_amdgcn_global_load_async_to_lds_b128(
        (v4i*)const_cast<unsigned short*>(g), (v4i_lds*)lds, 0, 0);
#else
    *reinterpret_cast<uint4*>(lds) = *reinterpret_cast<const uint4*>(g);
#endif
}

__device__ __forceinline__ void wait_async0() {
#ifdef HAVE_ASYNC_LDS
#  if __has_builtin(__builtin_amdgcn_s_wait_asynccnt)
    __builtin_amdgcn_s_wait_asynccnt(0);
#  else
    asm volatile("s_wait_asynccnt 0" ::: "memory");
#  endif
#endif
}

// ---------------- elementwise convert kernels ----------------
__global__ void cvt_bf16_kernel(const float* __restrict__ src, unsigned short* __restrict__ dst,
                                long long n) {
    long long i = (long long)blockIdx.x * blockDim.x + threadIdx.x;
    if (i < n) dst[i] = f2bf(src[i]);
}

__global__ void transpose_cvt_kernel(const float* __restrict__ W, unsigned short* __restrict__ WT,
                                     int K, int Nn) {
    long long i = (long long)blockIdx.x * blockDim.x + threadIdx.x;
    long long tot = (long long)K * Nn;
    if (i >= tot) return;
    int n = (int)(i % Nn);
    int k = (int)(i / Nn);
    WT[(size_t)n * K + k] = f2bf(W[i]);
}

// RoPE + convert to bf16. src layout [b][n][h*64+d] (fp32), dst same layout bf16.
__global__ void rope_cvt_kernel(const float* __restrict__ src, unsigned short* __restrict__ dst,
                                const float* __restrict__ cosb, const float* __restrict__ sinb,
                                int nheads, long long total) {
    long long i = (long long)blockIdx.x * blockDim.x + threadIdx.x;
    if (i >= total) return;
    int d = (int)(i & 63);
    long long row = i >> 6;            // (b*N + n)*nheads + h
    long long bn  = row / nheads;      // b*N + n
    int n = (int)(bn % NSEQ);
    float c = cosb[n * HDIM + d];
    float s = sinb[n * HDIM + d];
    float x = src[i];
    float other = (d < 32) ? -src[i + 32] : src[i - 32];
    dst[i] = f2bf(x * c + other * s);
}

// ---------------- LDS-tiled WMMA bf16 GEMM ----------------
// C[M,Nn] = A[M,K] * B[K,Nn], B supplied transposed (BT[Nn,K], row-major).
// Workgroup: 256 threads = 8 waves; macro tile TM x TN (128x64); wave tile 32x32 (2x2 WMMA tiles).
// LDS tiles double-buffered and filled with async global->LDS b128 copies.
__global__ __launch_bounds__(256) void gemm_tiled_wmma(const unsigned short* __restrict__ A,
                                                       const unsigned short* __restrict__ BT,
                                                       float* __restrict__ Cf,
                                                       unsigned short* __restrict__ Cb,
                                                       int M, int Nn, int K) {
    __shared__ unsigned short Asm[2][TM * KC];   // 2 x 8 KB
    __shared__ unsigned short Bsm[2][TN * KC];   // 2 x 4 KB
    const int tid  = threadIdx.x;
    const int lane = tid & 31;
    const int wv   = tid >> 5;
    const int lm   = lane & 15, half = lane >> 4;

    const int tilesN = Nn / TN;
    const int bm = (blockIdx.x / tilesN) * TM;
    const int bn = (blockIdx.x % tilesN) * TN;
    const int wm = (wv & 3) * 32;    // wave M offset inside macro tile
    const int wn = (wv >> 2) * 32;   // wave N offset inside macro tile

    floatx8 acc[2][2] = {};

    auto issue_tiles = [&](int buf, int kk) {
        // A tile: TM x KC bf16 = 8192 B = 512 x 16B chunks, 2 per thread
#pragma unroll
        for (int c = 0; c < 2; ++c) {
            int ch = tid + c * 256;
            int r = ch >> 2, cc = ch & 3;
            cp16_to_lds(&Asm[buf][r * KC + cc * 8],
                        A + (size_t)(bm + r) * K + kk + cc * 8);
        }
        // B tile: TN x KC bf16 = 4096 B = 256 x 16B chunks, 1 per thread
        {
            int r = tid >> 2, cc = tid & 3;
            cp16_to_lds(&Bsm[buf][r * KC + cc * 8],
                        BT + (size_t)(bn + r) * K + kk + cc * 8);
        }
    };

    issue_tiles(0, 0);
    wait_async0();
    __syncthreads();

    const int nk = K / KC;
    int buf = 0;
    for (int ks = 0; ks < nk; ++ks) {
        int nxt = buf ^ 1;
        if (ks + 1 < nk) issue_tiles(nxt, (ks + 1) * KC);   // async prefetch next tiles

        Frag af[2], bfr[2];
#pragma unroll
        for (int i = 0; i < 2; ++i)
            load_a_frag(af[i], &Asm[buf][(wm + i * 16 + lm) * KC], half);
#pragma unroll
        for (int j = 0; j < 2; ++j)
            load_bt_frag(bfr[j], &Bsm[buf][(wn + j * 16 + lm) * KC], half);
#pragma unroll
        for (int i = 0; i < 2; ++i)
#pragma unroll
            for (int j = 0; j < 2; ++j)
                acc[i][j] = __builtin_amdgcn_wmma_f32_16x16x32_bf16(
                    false, af[i].v, false, bfr[j].v, (short)0, acc[i][j], false, false);

        wait_async0();      // prefetch into nxt complete
        __syncthreads();    // everyone done reading buf before it is overwritten
        buf = nxt;
    }

    // C/D layout: lane holds col n=lm, rows m = r + 8*half
#pragma unroll
    for (int i = 0; i < 2; ++i) {
#pragma unroll
        for (int j = 0; j < 2; ++j) {
#pragma unroll
            for (int r = 0; r < 8; ++r) {
                int m = bm + wm + i * 16 + r + 8 * half;
                int n = bn + wn + j * 16 + lm;
                float vv = acc[i][j][r];
                if (Cf) Cf[(size_t)m * Nn + n] = vv;
                else    Cb[(size_t)m * Nn + n] = f2bf(vv);
            }
        }
    }
}

// ---------------- Flash attention: one wave per (b, head, 16-query tile) ----------------
__global__ void attn_wmma(const unsigned short* __restrict__ Qb,
                          const unsigned short* __restrict__ Kb,
                          const unsigned short* __restrict__ Vb,
                          unsigned short* __restrict__ Ctx) {
    __shared__ unsigned short Plds[8 * 16 * 32];   // per-wave 16x32 bf16 P tile
    int wv   = threadIdx.x >> 5;
    int lane = threadIdx.x & 31;
    int wg = blockIdx.x * (blockDim.x >> 5) + wv;
    int qt = wg & 127;            // N/16 = 128
    int h  = (wg >> 7) & 31;
    int b  = wg >> 12;
    int g  = h >> 2;              // R = 4 heads per group
    int lm = lane & 15, half = lane >> 4;
    int qbase = qt << 4;

    const unsigned short* qrow = Qb + (size_t)(b * NSEQ + qbase + lm) * DOUT + h * HDIM;
    Frag qa0, qa1;
    load_a_frag(qa0, qrow, half);
    load_a_frag(qa1, qrow + 32, half);

    float m_i[8], l_i[8];
    floatx8 Oacc[4] = {};
#pragma unroll
    for (int r = 0; r < 8; ++r) { m_i[r] = -1e30f; l_i[r] = 0.0f; }

    unsigned short* pl = Plds + wv * 512;
    int jmax = (qbase + 15) >> 5;
    for (int j = 0; j <= jmax; ++j) {
        int kbase = j << 5;
        floatx8 S[2];
#pragma unroll
        for (int t = 0; t < 2; ++t) {
            const unsigned short* krow =
                Kb + (size_t)(b * NSEQ + kbase + 16 * t + lm) * KVD + g * HDIM;
            Frag kb0, kb1;
            load_bt_frag(kb0, krow, half);
            load_bt_frag(kb1, krow + 32, half);
            floatx8 acc = {};
            acc = __builtin_amdgcn_wmma_f32_16x16x32_bf16(false, qa0.v, false, kb0.v,
                                                          (short)0, acc, false, false);
            acc = __builtin_amdgcn_wmma_f32_16x16x32_bf16(false, qa1.v, false, kb1.v,
                                                          (short)0, acc, false, false);
            S[t] = acc;
        }
        float bm[8];
#pragma unroll
        for (int r = 0; r < 8; ++r) {
            int qrowi = qbase + r + 8 * half;
            float s0 = S[0][r] * 0.125f;          // 1/sqrt(64)
            float s1 = S[1][r] * 0.125f;
            if (kbase + lm      > qrowi) s0 = -1e30f;
            if (kbase + 16 + lm > qrowi) s1 = -1e30f;
            S[0][r] = s0; S[1][r] = s1;
            bm[r] = fmaxf(s0, s1);
        }
#pragma unroll
        for (int r = 0; r < 8; ++r) {
            float v = bm[r];
            v = fmaxf(v, __shfl_xor(v, 1, 16));
            v = fmaxf(v, __shfl_xor(v, 2, 16));
            v = fmaxf(v, __shfl_xor(v, 4, 16));
            v = fmaxf(v, __shfl_xor(v, 8, 16));
            bm[r] = v;
        }
        float alpha[8];
#pragma unroll
        for (int r = 0; r < 8; ++r) {
            float mn = fmaxf(m_i[r], bm[r]);
            alpha[r] = __expf(m_i[r] - mn);
            m_i[r] = mn;
            float p0 = __expf(S[0][r] - mn);
            float p1 = __expf(S[1][r] - mn);
            float rs = p0 + p1;
            rs += __shfl_xor(rs, 1, 16);
            rs += __shfl_xor(rs, 2, 16);
            rs += __shfl_xor(rs, 4, 16);
            rs += __shfl_xor(rs, 8, 16);
            l_i[r] = l_i[r] * alpha[r] + rs;
            pl[(r + 8 * half) * 32 + lm]      = f2bf(p0);
            pl[(r + 8 * half) * 32 + 16 + lm] = f2bf(p1);
        }
#pragma unroll
        for (int td = 0; td < 4; ++td)
#pragma unroll
            for (int r = 0; r < 8; ++r) Oacc[td][r] *= alpha[r];

        Frag pf;
        {
            const unsigned int* lp = reinterpret_cast<const unsigned int*>(pl + lm * 32);
#pragma unroll
            for (int p = 0; p < 8; ++p) pf.u[p] = lp[p + 8 * half];
        }
#pragma unroll
        for (int td = 0; td < 4; ++td) {
            Frag vf;
#pragma unroll
            for (int e = 0; e < 16; ++e) {
                int key = kbase + e + 16 * half;
                vf.s[e] = Vb[(size_t)(b * NSEQ + key) * KVD + g * HDIM + td * 16 + lm];
            }
            Oacc[td] = __builtin_amdgcn_wmma_f32_16x16x32_bf16(false, pf.v, false, vf.v,
                                                               (short)0, Oacc[td], false, false);
        }
    }
#pragma unroll
    for (int td = 0; td < 4; ++td) {
#pragma unroll
        for (int r = 0; r < 8; ++r) {
            int m = qbase + r + 8 * half;
            float o = Oacc[td][r] / l_i[r];
            Ctx[(size_t)(b * NSEQ + m) * DOUT + h * HDIM + td * 16 + lm] = f2bf(o);
        }
    }
}

// ---------------- launch ----------------
extern "C" void kernel_launch(void* const* d_in, const int* in_sizes, int n_in,
                              void* d_out, int out_size, void* d_ws, size_t ws_size,
                              hipStream_t stream) {
    const float* x    = (const float*)d_in[0];
    const float* cosb = (const float*)d_in[1];
    const float* sinb = (const float*)d_in[2];
    // d_in[3] = causal mask (recomputed analytically, unused)
    const float* Wq = (const float*)d_in[4];
    const float* Wk = (const float*)d_in[5];
    const float* Wv = (const float*)d_in[6];
    const float* Wo = (const float*)d_in[7];
    float* out = (float*)d_out;

    char* ws = (char*)d_ws;
    const size_t MB = 1024 * 1024;
    unsigned short* xb  = (unsigned short*)(ws + 0);        // 16 MB  x bf16
    unsigned short* WqT = (unsigned short*)(ws + 16 * MB);  //  8 MB  Wq^T bf16
    unsigned short* WkT = (unsigned short*)(ws + 24 * MB);  //  2 MB
    unsigned short* WvT = (unsigned short*)(ws + 26 * MB);  //  2 MB
    unsigned short* WoT = (unsigned short*)(ws + 28 * MB);  //  8 MB
    unsigned short* Qbf = (unsigned short*)(ws + 36 * MB);  // 16 MB  Q (post-RoPE) bf16
    unsigned short* Kbf = (unsigned short*)(ws + 52 * MB);  //  4 MB  K (post-RoPE) bf16
    unsigned short* Vbf = (unsigned short*)(ws + 56 * MB);  //  4 MB  V bf16
    float*          Qf  = (float*)(ws + 60 * MB);           // 32 MB  Q fp32 scratch
    unsigned short* Ctx = (unsigned short*)(ws + 60 * MB);  // 16 MB  aliases Qf (dead by then)
    float*          Kf  = (float*)(ws + 92 * MB);           //  8 MB  K fp32 scratch
    (void)in_sizes; (void)n_in; (void)out_size; (void)ws_size; // 100 MB total

    const int THR = 256;
    // 1) convert x and weights
    {
        long long nx = (long long)BB * NSEQ * DIN;   // 8M
        cvt_bf16_kernel<<<(unsigned)((nx + THR - 1) / THR), THR, 0, stream>>>(x, xb, nx);
        long long nq = (long long)DIN * DOUT;        // 4M
        transpose_cvt_kernel<<<(unsigned)((nq + THR - 1) / THR), THR, 0, stream>>>(Wq, WqT, DIN, DOUT);
        long long nk = (long long)DIN * KVD;         // 1M
        transpose_cvt_kernel<<<(unsigned)((nk + THR - 1) / THR), THR, 0, stream>>>(Wk, WkT, DIN, KVD);
        transpose_cvt_kernel<<<(unsigned)((nk + THR - 1) / THR), THR, 0, stream>>>(Wv, WvT, DIN, KVD);
        transpose_cvt_kernel<<<(unsigned)((nq + THR - 1) / THR), THR, 0, stream>>>(Wo, WoT, DOUT, DOUT);
    }
    const int M = BB * NSEQ;   // 4096
    // 2) QKV projections: LDS-tiled WMMA GEMM, one 128x64 macro tile per workgroup
    {
        int blkQ = (M / TM) * (DOUT / TN);           // 32*32 = 1024
        gemm_tiled_wmma<<<blkQ, THR, 0, stream>>>(xb, WqT, Qf, nullptr, M, DOUT, DIN);
        int blkK = (M / TM) * (KVD / TN);            // 32*8 = 256
        gemm_tiled_wmma<<<blkK, THR, 0, stream>>>(xb, WkT, Kf, nullptr, M, KVD, DIN);
        gemm_tiled_wmma<<<blkK, THR, 0, stream>>>(xb, WvT, nullptr, Vbf, M, KVD, DIN);
    }
    // 3) RoPE + bf16 convert
    {
        long long tq = (long long)BB * NSEQ * NH * HDIM;   // 8M
        rope_cvt_kernel<<<(unsigned)((tq + THR - 1) / THR), THR, 0, stream>>>(Qf, Qbf, cosb, sinb, NH, tq);
        long long tk = (long long)BB * NSEQ * NG * HDIM;   // 2M
        rope_cvt_kernel<<<(unsigned)((tk + THR - 1) / THR), THR, 0, stream>>>(Kf, Kbf, cosb, sinb, NG, tk);
    }
    // 4) flash attention: B*H*(N/16) = 8192 waves -> 1024 blocks of 8 waves
    attn_wmma<<<(BB * NH * (NSEQ / 16)) / 8, THR, 0, stream>>>(Qbf, Kbf, Vbf, Ctx);
    // 5) output projection -> fp32 d_out
    {
        int blkO = (M / TM) * (DOUT / TN);
        gemm_tiled_wmma<<<blkO, THR, 0, stream>>>(Ctx, WoT, out, nullptr, M, DOUT, DIN);
    }
}
